// RealityStoneGPT2_79989470921186
// MI455X (gfx1250) — compile-verified, matrix operand order checked by reading, not run on hardware
//
#include <hip/hip_runtime.h>

// ---------------------------------------------------------------- constants
static constexpr int BB = 2;        // batch
static constexpr int SS = 1024;     // seq
static constexpr int DD = 1024;     // model dim
static constexpr int HH = 16;       // heads
static constexpr int LL = 6;        // layers
static constexpr int DHq = 64;      // head dim
static constexpr int D3 = 3 * DD;   // 3072
static constexpr int D4 = 4 * DD;   // 4096

#define Cc   0.1f
#define SCc  0.31622776601683794f   // sqrt(0.1)
#define LNE  1e-5f

// ---------------------------------------------------------------- types
typedef __bf16 bf16;
typedef __attribute__((ext_vector_type(8)))  bf16  bf16x8;
typedef __attribute__((ext_vector_type(16))) bf16  bf16x16;
typedef __attribute__((ext_vector_type(8)))  float floatx8;
typedef __attribute__((ext_vector_type(4)))  int   v4i;

// ---------------------------------------------------------------- async-to-LDS (CDNA5)
#if defined(__has_builtin)
#if __has_builtin(__builtin_amdgcn_global_load_async_to_lds_b128)
#define HAVE_ASYNC_LDS 1
#endif
#endif

#if defined(HAVE_ASYNC_LDS)
__device__ __forceinline__ void async_copy_b128(const bf16* g, bf16* l) {
    __builtin_amdgcn_global_load_async_to_lds_b128(
        (v4i*)(void*)(const_cast<bf16*>(g)), (v4i*)(void*)l, 0, 0);
}
__device__ __forceinline__ void wait_async0() {
#if __has_builtin(__builtin_amdgcn_s_wait_asynccnt)
    __builtin_amdgcn_s_wait_asynccnt(0);
#else
    asm volatile("s_wait_asynccnt 0x0" ::: "memory");
#endif
}
#endif

// ---------------------------------------------------------------- helpers
__device__ __forceinline__ float blockSum(float v, float* red) {
    __syncthreads();
    #pragma unroll
    for (int o = 16; o > 0; o >>= 1) v += __shfl_down(v, o, 32);
    if ((threadIdx.x & 31) == 0) red[threadIdx.x >> 5] = v;
    __syncthreads();
    if (threadIdx.x == 0) {
        float s = 0.f;
        #pragma unroll
        for (int i = 0; i < 8; ++i) s += red[i];
        red[0] = s;
    }
    __syncthreads();
    return red[0];
}

__device__ __forceinline__ float blockMax(float v, float* red) {
    __syncthreads();
    #pragma unroll
    for (int o = 16; o > 0; o >>= 1) v = fmaxf(v, __shfl_down(v, o, 32));
    if ((threadIdx.x & 31) == 0) red[threadIdx.x >> 5] = v;
    __syncthreads();
    if (threadIdx.x == 0) {
        float s = -1e30f;
        #pragma unroll
        for (int i = 0; i < 8; ++i) s = fmaxf(s, red[i]);
        red[0] = s;
    }
    __syncthreads();
    return red[0];
}

__device__ __forceinline__ float log0_scale(float n2) {
    float n = fmaxf(sqrtf(n2), 1e-10f);
    float arg = SCc * fminf(n, 1.0f - 1e-7f);
    return atanhf(arg) / (SCc * n);
}
__device__ __forceinline__ float exp0_scale(float n2) {
    float n = fmaxf(sqrtf(n2), 1e-10f);
    return tanhf(SCc * n) / (SCc * n);
}

// ================================================================ GEMM
// C[M,N] = A[M,K](bf16 row-major) * Bt[N,K](bf16 row-major) + bias[N]
// Block tile BMxBN, WM*WN waves, wave tile (MT*16)x(NT*16) of 16x16x32 WMMAs.
// Double-buffered LDS, one barrier per K-step; A/B staged either by
// GLOBAL_LOAD_ASYNC_TO_LDS_B128 (ASYNCcnt) or VGPR-staged b128 loads.
template<int BM, int BN, int WM, int WN>
__global__ __launch_bounds__(WM * WN * 32) void gemm_bf16_wmma(
    const bf16* __restrict__ A, const bf16* __restrict__ Bt,
    const float* __restrict__ bias, void* __restrict__ Cout,
    int K, int lda, int ldb, int ldc,
    long long aOffO, long long aOffI, long long bOffO, long long bOffI,
    long long cOffO, long long cOffI, int innerCount, int outBf16)
{
    constexpr int THREADS = WM * WN * 32;
    constexpr int MT = BM / (WM * 16);
    constexpr int NT = BN / (WN * 16);
    constexpr int LR = 48;                           // padded LDS row (bf16)
    static_assert(BM * 2 == THREADS && BN * 2 == THREADS, "loader mapping");

    __shared__ __align__(16) bf16 As[2][BM * LR];
    __shared__ __align__(16) bf16 Bs[2][BN * LR];

    const int z  = blockIdx.z;
    const int zo = z / innerCount;
    const int zi = z % innerCount;
    A  += (long long)zo * aOffO + (long long)zi * aOffI;
    Bt += (long long)zo * bOffO + (long long)zi * bOffI;
    const long long cbase = (long long)zo * cOffO + (long long)zi * cOffI;

    const int tid  = threadIdx.x;
    const int wave = tid >> 5;
    const int lane = tid & 31;
    const int wm0  = (wave / WN) * (MT * 16);
    const int wn0  = (wave % WN) * (NT * 16);
    const int tileM = blockIdx.y * BM;
    const int tileN = blockIdx.x * BN;
    const int half = lane >> 4, l16 = lane & 15, kofs = half * 8;

    // loader: one 16-elem (2x b128) chunk per thread per matrix
    const int crow = tid >> 1;
    const int ccol = (tid & 1) * 16;
    const bf16* aSrc = A  + (long long)(tileM + crow) * lda + ccol;
    const bf16* bSrc = Bt + (long long)(tileN + crow) * ldb + ccol;
    const int ldsOff = crow * LR + ccol;

    floatx8 acc[MT][NT] = {};

    // ---- prologue: stage K-tile 0 into buffer 0
#if defined(HAVE_ASYNC_LDS)
    async_copy_b128(aSrc,     &As[0][ldsOff]);
    async_copy_b128(aSrc + 8, &As[0][ldsOff + 8]);
    async_copy_b128(bSrc,     &Bs[0][ldsOff]);
    async_copy_b128(bSrc + 8, &Bs[0][ldsOff + 8]);
#else
    {
        bf16x8 a0 = *(const bf16x8*)(aSrc);
        bf16x8 a1 = *(const bf16x8*)(aSrc + 8);
        bf16x8 b0 = *(const bf16x8*)(bSrc);
        bf16x8 b1 = *(const bf16x8*)(bSrc + 8);
        *(bf16x8*)(&As[0][ldsOff])     = a0;
        *(bf16x8*)(&As[0][ldsOff + 8]) = a1;
        *(bf16x8*)(&Bs[0][ldsOff])     = b0;
        *(bf16x8*)(&Bs[0][ldsOff + 8]) = b1;
    }
#endif

    int cur = 0;
    for (int k0 = 0; k0 < K; k0 += 32) {
#if defined(HAVE_ASYNC_LDS)
        wait_async0();           // own DMA landed ...
#endif
        __syncthreads();         // ... then everyone's (per-wave dscnt/async fenced)

        const bool more = (k0 + 32 < K);
        const int  nxt  = cur ^ 1;
#if defined(HAVE_ASYNC_LDS)
        if (more) {
            const bf16* an = aSrc + k0 + 32;
            const bf16* bn = bSrc + k0 + 32;
            async_copy_b128(an,     &As[nxt][ldsOff]);
            async_copy_b128(an + 8, &As[nxt][ldsOff + 8]);
            async_copy_b128(bn,     &Bs[nxt][ldsOff]);
            async_copy_b128(bn + 8, &Bs[nxt][ldsOff + 8]);
        }
#else
        bf16x8 ra0, ra1, rb0, rb1;
        if (more) {
            const bf16* an = aSrc + k0 + 32;
            const bf16* bn = bSrc + k0 + 32;
            ra0 = *(const bf16x8*)(an);
            ra1 = *(const bf16x8*)(an + 8);
            rb0 = *(const bf16x8*)(bn);
            rb1 = *(const bf16x8*)(bn + 8);
            if (k0 + 64 < K) {
                __builtin_prefetch(an + 32, 0, 3);
                __builtin_prefetch(bn + 32, 0, 3);
            }
        }
#endif
        // ---- compute from buffer `cur`
        bf16x16 afr[MT], bfr[NT];
        #pragma unroll
        for (int i = 0; i < MT; ++i) {
            const bf16* ap = &As[cur][(wm0 + i * 16 + l16) * LR + kofs];
            bf16x8 lo = *(const bf16x8*)(ap);
            bf16x8 hi = *(const bf16x8*)(ap + 16);
            afr[i] = __builtin_shufflevector(lo, hi,
                     0,1,2,3,4,5,6,7,8,9,10,11,12,13,14,15);
        }
        #pragma unroll
        for (int j = 0; j < NT; ++j) {
            const bf16* bp = &Bs[cur][(wn0 + j * 16 + l16) * LR + kofs];
            bf16x8 lo = *(const bf16x8*)(bp);
            bf16x8 hi = *(const bf16x8*)(bp + 16);
            bfr[j] = __builtin_shufflevector(lo, hi,
                     0,1,2,3,4,5,6,7,8,9,10,11,12,13,14,15);
        }
        #pragma unroll
        for (int i = 0; i < MT; ++i)
            #pragma unroll
            for (int j = 0; j < NT; ++j)
                acc[i][j] = __builtin_amdgcn_wmma_f32_16x16x32_bf16(
                    false, afr[i], false, bfr[j],
                    (short)0, acc[i][j], false, false);

#if !defined(HAVE_ASYNC_LDS)
        if (more) {
            *(bf16x8*)(&As[nxt][ldsOff])     = ra0;
            *(bf16x8*)(&As[nxt][ldsOff + 8]) = ra1;
            *(bf16x8*)(&Bs[nxt][ldsOff])     = rb0;
            *(bf16x8*)(&Bs[nxt][ldsOff + 8]) = rb1;
        }
#endif
        cur = nxt;
    }

    // ---- epilogue: C layout — vgpr r: lanes 0-15 M=r, lanes 16-31 M=r+8; N=lane%16
    #pragma unroll
    for (int i = 0; i < MT; ++i) {
        #pragma unroll
        for (int j = 0; j < NT; ++j) {
            const int n = tileN + wn0 + j * 16 + l16;
            const float bv = bias ? bias[n] : 0.f;
            #pragma unroll
            for (int r = 0; r < 8; ++r) {
                const int m = tileM + wm0 + i * 16 + half * 8 + r;
                const float v = acc[i][j][r] + bv;
                const long long idx = cbase + (long long)m * ldc + n;
                if (outBf16) ((bf16*)Cout)[idx] = (bf16)v;
                else         ((float*)Cout)[idx] = v;
            }
        }
    }
}

// ================================================================ transposes / converts
__global__ __launch_bounds__(256) void k_transpose_w(
    const float* __restrict__ W, bf16* __restrict__ Wt, int K, int N)
{
    __shared__ float tile[32][33];
    const long long zoff = (long long)blockIdx.z * (long long)K * N;
    const float* Wz  = W  + zoff;
    bf16*        Wtz = Wt + zoff;
    const int k0 = blockIdx.y * 32, n0 = blockIdx.x * 32;
    const int tx = threadIdx.x & 31, ty = threadIdx.x >> 5;  // 32x8
    #pragma unroll
    for (int i = 0; i < 32; i += 8)
        tile[ty + i][tx] = Wz[(long long)(k0 + ty + i) * N + n0 + tx];
    __syncthreads();
    #pragma unroll
    for (int i = 0; i < 32; i += 8)
        Wtz[(long long)(n0 + ty + i) * K + k0 + tx] = (bf16)tile[tx][ty + i];
}

__global__ void k_f32_to_bf16(const float* __restrict__ in, bf16* __restrict__ out, long long n) {
    long long i = (long long)blockIdx.x * blockDim.x + threadIdx.x;
    const long long stride = (long long)gridDim.x * blockDim.x;
    for (; i < n; i += stride) out[i] = (bf16)in[i];
}

// vt[b][h][d][s] = qkv[b][s][2D + h*DH + d]
__global__ void k_build_vt(const float* __restrict__ qkv, bf16* __restrict__ vt) {
    const long long total = (long long)BB * HH * DHq * SS;
    long long i = (long long)blockIdx.x * blockDim.x + threadIdx.x;
    if (i >= total) return;
    const int s = (int)(i % SS);
    long long t = i / SS;
    const int d = (int)(t % DHq); t /= DHq;
    const int h = (int)(t % HH);
    const int b = (int)(t / HH);
    vt[i] = (bf16)qkv[((long long)(b * SS + s)) * D3 + 2 * DD + h * DHq + d];
}

// ================================================================ row kernels
__global__ __launch_bounds__(256) void k_embed(
    const int* __restrict__ ids, const float* __restrict__ wte,
    const float* __restrict__ wpe, float* __restrict__ h)
{
    __shared__ float es[DD];
    __shared__ float red[8];
    const int row = blockIdx.x;                   // [0, B*S)
    const int s = row % SS;
    const long long wt = (long long)ids[row] * DD;
    const long long wp = (long long)s * DD;
    float s2 = 0.f;
    for (int i = threadIdx.x; i < DD; i += 256) {
        const float e = wte[wt + i] + wpe[wp + i];
        es[i] = e; s2 += e * e;
    }
    const float n2 = blockSum(s2, red);
    const float a = exp0_scale(n2);
    const long long r0 = (long long)row * DD;
    for (int i = threadIdx.x; i < DD; i += 256) h[r0 + i] = a * es[i];
}

// t = LN(log0(h)) * g + b   (bf16 out) — log0 is a row-scalar scale
__global__ __launch_bounds__(256) void k_ln_log0(
    const float* __restrict__ h, bf16* __restrict__ t,
    const float* __restrict__ g, const float* __restrict__ bb)
{
    __shared__ float row[DD];
    __shared__ float red[8];
    const long long r0 = (long long)blockIdx.x * DD;
    float s1 = 0.f, s2 = 0.f;
    for (int i = threadIdx.x; i < DD; i += 256) {
        const float x = h[r0 + i];
        row[i] = x; s1 += x; s2 += x * x;
    }
    const float sum   = blockSum(s1, red);
    const float sumsq = blockSum(s2, red);
    const float sig = log0_scale(sumsq);
    const float mu  = sig * sum / (float)DD;
    const float ex2 = sig * sig * sumsq / (float)DD;
    const float inv = rsqrtf(ex2 - mu * mu + LNE);
    for (int i = threadIdx.x; i < DD; i += 256)
        t[r0 + i] = (bf16)(((sig * row[i] - mu) * inv) * g[i] + bb[i]);
}

// scores row: p = softmax(s/8 + mask)   (in-place bf16)
__global__ __launch_bounds__(256) void k_softmax(
    bf16* __restrict__ sc, const float* __restrict__ am)
{
    __shared__ float vals[SS];
    __shared__ float red[8];
    const long long row = blockIdx.x;             // [0, B*H*S)
    const int b = (int)(row / ((long long)HH * SS));
    const long long r0 = row * SS;
    float mx = -1e30f;
    for (int i = threadIdx.x; i < SS; i += 256) {
        const float v = (float)sc[r0 + i] * 0.125f
                      + (1.f - am[b * SS + i]) * -10000.f;
        vals[i] = v; mx = fmaxf(mx, v);
    }
    mx = blockMax(mx, red);
    float se = 0.f;
    for (int i = threadIdx.x; i < SS; i += 256) {
        const float e = __expf(vals[i] - mx);
        vals[i] = e; se += e;
    }
    const float s = blockSum(se, red);
    const float inv = 1.f / s;
    for (int i = threadIdx.x; i < SS; i += 256)
        sc[r0 + i] = (bf16)(vals[i] * inv);
}

// h = mobius_add(h, exp0(r)) — scalar coefficients from 3 row dots
__global__ __launch_bounds__(256) void k_residual(
    float* __restrict__ h, const float* __restrict__ r)
{
    __shared__ float hs[DD], rs[DD];
    __shared__ float red[8];
    const long long r0 = (long long)blockIdx.x * DD;
    float shh = 0.f, srr = 0.f, shr = 0.f;
    for (int i = threadIdx.x; i < DD; i += 256) {
        const float x = h[r0 + i], y = r[r0 + i];
        hs[i] = x; rs[i] = y;
        shh += x * x; srr += y * y; shr += x * y;
    }
    const float hh = blockSum(shh, red);
    const float rr = blockSum(srr, red);
    const float hr = blockSum(shr, red);
    const float a  = exp0_scale(rr);
    const float x2 = hh, y2 = a * a * rr, xy = a * hr;
    const float den = fmaxf(1.f + 2.f * Cc * xy + Cc * Cc * x2 * y2, 1e-10f);
    const float ch = (1.f + 2.f * Cc * xy + Cc * y2) / den;
    const float cr = (1.f - Cc * x2) * a / den;
    for (int i = threadIdx.x; i < DD; i += 256)
        h[r0 + i] = ch * hs[i] + cr * rs[i];
}

// o = log0( pball( exp0(z), exp0(gelu(z)), 0.7 ) )  — fully analytic scalars
__global__ __launch_bounds__(256) void k_ffn_act(
    const float* __restrict__ z, bf16* __restrict__ o)
{
    __shared__ float zs[D4], gs[D4];
    __shared__ float red[8];
    const long long r0 = (long long)blockIdx.x * D4;
    float szz = 0.f, sgg = 0.f, szg = 0.f;
    for (int i = threadIdx.x; i < D4; i += 256) {
        const float x = z[r0 + i];
        const float gg = 0.5f * x * (1.f + erff(x * 0.70710678118f));
        zs[i] = x; gs[i] = gg;
        szz += x * x; sgg += gg * gg; szg += x * gg;
    }
    const float zz  = blockSum(szz, red);
    const float gg2 = blockSum(sgg, red);
    const float zg  = blockSum(szg, red);
    const float nz = fmaxf(sqrtf(zz),  1e-10f);
    const float ng = fmaxf(sqrtf(gg2), 1e-10f);
    const float b1 = tanhf(0.3f * SCc * nz) / (SCc * nz);
    const float b2 = tanhf(0.7f * SCc * ng) / (SCc * ng);
    const float x2 = b1 * b1 * zz, y2 = b2 * b2 * gg2, xy = b1 * b2 * zg;
    const float den = fmaxf(1.f + 2.f * Cc * xy + Cc * Cc * x2 * y2, 1e-10f);
    const float Az = (1.f + 2.f * Cc * xy + Cc * y2) * b1 / den;
    const float Bg = (1.f - Cc * x2) * b2 / den;
    const float m2 = Az * Az * zz + 2.f * Az * Bg * zg + Bg * Bg * gg2;
    const float sig = log0_scale(m2);
    for (int i = threadIdx.x; i < D4; i += 256)
        o[r0 + i] = (bf16)(sig * (Az * zs[i] + Bg * gs[i]));
}

// out = exp0( LN(log0(h)) * g + b )
__global__ __launch_bounds__(256) void k_final(
    const float* __restrict__ h, const float* __restrict__ g,
    const float* __restrict__ bb, float* __restrict__ out)
{
    __shared__ float hs[DD];
    __shared__ float red[8];
    const long long r0 = (long long)blockIdx.x * DD;
    float s1 = 0.f, s2 = 0.f;
    for (int i = threadIdx.x; i < DD; i += 256) {
        const float x = h[r0 + i];
        hs[i] = x; s1 += x; s2 += x * x;
    }
    const float sum   = blockSum(s1, red);
    const float sumsq = blockSum(s2, red);
    const float sig = log0_scale(sumsq);
    const float mu  = sig * sum / (float)DD;
    const float ex2 = sig * sig * sumsq / (float)DD;
    const float inv = rsqrtf(ex2 - mu * mu + LNE);
    float sy2 = 0.f;
    for (int i = threadIdx.x; i < DD; i += 256) {
        const float y = ((sig * hs[i] - mu) * inv) * g[i] + bb[i];
        sy2 += y * y;
    }
    const float yy = blockSum(sy2, red);
    const float a2 = exp0_scale(yy);
    for (int i = threadIdx.x; i < DD; i += 256) {
        const float y = ((sig * hs[i] - mu) * inv) * g[i] + bb[i];
        out[r0 + i] = a2 * y;
    }
}

// ================================================================ host
template<int BM, int BN, int WM, int WN>
static inline void launch_gemm(const bf16* A, const bf16* Bt, const float* bias,
    void* C, int M, int N, int K, int lda, int ldb, int ldc,
    long long aO, long long aI, long long bO, long long bI,
    long long cO, long long cI, int inner, int batches, int outBf16,
    hipStream_t st)
{
    dim3 grid(N / BN, M / BM, batches), block(WM * WN * 32);
    gemm_bf16_wmma<BM, BN, WM, WN><<<grid, block, 0, st>>>(
        A, Bt, bias, C, K, lda, ldb, ldc, aO, aI, bO, bI, cO, cI, inner, outBf16);
}

extern "C" void kernel_launch(void* const* d_in, const int* in_sizes, int n_in,
                              void* d_out, int out_size, void* d_ws, size_t ws_size,
                              hipStream_t stream) {
    (void)in_sizes; (void)n_in; (void)out_size; (void)ws_size;
    const int*   ids   = (const int*)  d_in[0];
    const float* amask = (const float*)d_in[1];
    const float* wte   = (const float*)d_in[2];
    const float* wpe   = (const float*)d_in[3];
    const float* ln1g  = (const float*)d_in[4];
    const float* ln1b  = (const float*)d_in[5];
    const float* ln2g  = (const float*)d_in[6];
    const float* ln2b  = (const float*)d_in[7];
    const float* aw    = (const float*)d_in[8];
    const float* ab    = (const float*)d_in[9];
    const float* apw   = (const float*)d_in[10];
    const float* apb   = (const float*)d_in[11];
    const float* fcw   = (const float*)d_in[12];
    const float* fcb   = (const float*)d_in[13];
    const float* fpw   = (const float*)d_in[14];
    const float* fpb   = (const float*)d_in[15];
    const float* lnfg  = (const float*)d_in[16];
    const float* lnfb  = (const float*)d_in[17];
    float* out = (float*)d_out;

    // ------- workspace carve-up
    char* ws = (char*)d_ws;
    auto alloc = [&](size_t bytes) -> char* {
        char* p = ws;
        ws += (bytes + 255) & ~(size_t)255;
        return p;
    };
    const long long NW_ATT = (long long)LL * D3 * DD;
    const long long NW_PRJ = (long long)LL * DD * DD;
    const long long NW_FC  = (long long)LL * DD * D4;
    const long long NW_FP  = (long long)LL * D4 * DD;
    bf16*  wTa  = (bf16*) alloc(NW_ATT * 2);
    bf16*  wTp  = (bf16*) alloc(NW_PRJ * 2);
    bf16*  wTf  = (bf16*) alloc(NW_FC  * 2);
    bf16*  wTfp = (bf16*) alloc(NW_FP  * 2);
    const long long ROWS = (long long)BB * SS;          // 2048
    float* h    = (float*)alloc(ROWS * DD * 4);
    bf16*  tb   = (bf16*) alloc(ROWS * DD * 2);
    float* qkv  = (float*)alloc(ROWS * D3 * 4);
    bf16*  qkvb = (bf16*) alloc(ROWS * D3 * 2);
    bf16*  vt   = (bf16*) alloc((long long)BB * HH * DHq * SS * 2);
    bf16*  sc   = (bf16*) alloc((long long)BB * HH * SS * SS * 2);
    bf16*  aob  = (bf16*) alloc(ROWS * DD * 2);
    float* rbuf = (float*)alloc(ROWS * DD * 4);
    float* zbuf = (float*)alloc(ROWS * D4 * 4);
    bf16*  gbuf = (bf16*) alloc(ROWS * D4 * 2);

    // ------- weight prep: fp32 [K,N] -> bf16 [N,K], all layers at once
    {
        dim3 b(256);
        k_transpose_w<<<dim3(D3 / 32, DD / 32, LL), b, 0, stream>>>(aw,  wTa,  DD, D3);
        k_transpose_w<<<dim3(DD / 32, DD / 32, LL), b, 0, stream>>>(apw, wTp,  DD, DD);
        k_transpose_w<<<dim3(D4 / 32, DD / 32, LL), b, 0, stream>>>(fcw, wTf,  DD, D4);
        k_transpose_w<<<dim3(DD / 32, D4 / 32, LL), b, 0, stream>>>(fpw, wTfp, D4, DD);
    }

    // ------- embedding + exp0
    k_embed<<<dim3((int)ROWS), dim3(256), 0, stream>>>(ids, wte, wpe, h);

    const long long SD3 = (long long)SS * D3;
    for (int l = 0; l < LL; ++l) {
        const bf16* wTa_l  = wTa  + (long long)l * D3 * DD;
        const bf16* wTp_l  = wTp  + (long long)l * DD * DD;
        const bf16* wTf_l  = wTf  + (long long)l * DD * D4;
        const bf16* wTfp_l = wTfp + (long long)l * D4 * DD;

        // ---- attention
        k_ln_log0<<<dim3((int)ROWS), dim3(256), 0, stream>>>(h, tb,
                       ln1g + (long long)l * DD, ln1b + (long long)l * DD);
        launch_gemm<128, 128, 2, 4>(tb, wTa_l, ab + (long long)l * D3, qkv,
                    (int)ROWS, D3, DD, DD, DD, D3,
                    0, 0, 0, 0, 0, 0, 1, 1, /*outBf16=*/0, stream);
        k_f32_to_bf16<<<dim3(4096), dim3(256), 0, stream>>>(qkv, qkvb, ROWS * D3);
        {
            const long long nvt = (long long)BB * HH * DHq * SS;
            k_build_vt<<<dim3((int)((nvt + 255) / 256)), dim3(256), 0, stream>>>(qkv, vt);
        }
        // scores[b,h] = Q @ K^T
        launch_gemm<128, 128, 2, 4>(qkvb, qkvb + DD, nullptr, sc,
                    SS, SS, DHq, D3, D3, SS,
                    SD3, DHq, SD3, DHq,
                    (long long)HH * SS * SS, (long long)SS * SS,
                    HH, BB * HH, /*outBf16=*/1, stream);
        k_softmax<<<dim3(BB * HH * SS), dim3(256), 0, stream>>>(sc, amask);
        // ao[b,s,h*DH+d] = P @ V   (N = 64 -> 64x64 tile config)
        launch_gemm<64, 64, 2, 2>(sc, vt, nullptr, aob,
                    SS, DHq, SS, SS, SS, DD,
                    (long long)HH * SS * SS, (long long)SS * SS,
                    (long long)HH * DHq * SS, (long long)DHq * SS,
                    (long long)SS * DD, DHq,
                    HH, BB * HH, /*outBf16=*/1, stream);
        launch_gemm<128, 128, 2, 4>(aob, wTp_l, apb + (long long)l * DD, rbuf,
                    (int)ROWS, DD, DD, DD, DD, DD,
                    0, 0, 0, 0, 0, 0, 1, 1, /*outBf16=*/0, stream);
        k_residual<<<dim3((int)ROWS), dim3(256), 0, stream>>>(h, rbuf);

        // ---- FFN
        k_ln_log0<<<dim3((int)ROWS), dim3(256), 0, stream>>>(h, tb,
                       ln2g + (long long)l * DD, ln2b + (long long)l * DD);
        launch_gemm<128, 128, 2, 4>(tb, wTf_l, fcb + (long long)l * D4, zbuf,
                    (int)ROWS, D4, DD, DD, DD, D4,
                    0, 0, 0, 0, 0, 0, 1, 1, /*outBf16=*/0, stream);
        k_ffn_act<<<dim3((int)ROWS), dim3(256), 0, stream>>>(zbuf, gbuf);
        launch_gemm<128, 128, 2, 4>(gbuf, wTfp_l, fpb + (long long)l * DD, rbuf,
                    (int)ROWS, DD, D4, D4, D4, DD,
                    0, 0, 0, 0, 0, 0, 1, 1, /*outBf16=*/0, stream);
        k_residual<<<dim3((int)ROWS), dim3(256), 0, stream>>>(h, rbuf);
    }

    // ------- final LN + exp0
    k_final<<<dim3((int)ROWS), dim3(256), 0, stream>>>(h, lnfg, lnfb, out);
}
